// VehicleEncoder_40544491274512
// MI455X (gfx1250) — matrix operand order repaired and checked by reading, not compiled, not run on hardware
//
#include <hip/hip_runtime.h>
#include <stdint.h>

#define B_DIM   32
#define T_DIM   100
#define N_VEH   32
#define F_IN    8
#define D_DIM   256
#define LSTM_HD 512
#define GATE4   2048   // 4*LSTM_HD
#define ATT_OUT 256
#define LATENT  128

typedef _Float16 half8 __attribute__((ext_vector_type(8)));
typedef _Float16 v16h  __attribute__((ext_vector_type(16)));
typedef float    v8f   __attribute__((ext_vector_type(8)));

// ---------------- utility kernels ----------------
__global__ void k_zero_u32(uint32_t* p, int n) {
    int i = blockIdx.x * blockDim.x + threadIdx.x;
    if (i < n) p[i] = 0u;
}

__global__ void k_f32_to_f16(const float* __restrict__ s, _Float16* __restrict__ d, int n) {
    int i = blockIdx.x * blockDim.x + threadIdx.x;
    if (i < n) d[i] = (_Float16)s[i];
}

__global__ void k_bias_sum(const float* __restrict__ a, const float* __restrict__ b,
                           float* __restrict__ o, int n) {
    int i = blockIdx.x * blockDim.x + threadIdx.x;
    if (i < n) o[i] = a[i] + b[i];
}

// ---------------- GCN (collapsed): g_row[bt][d] = relu(mean_n(traj) @ gcn_w + b) ----
// One block (256 threads) per (b,t). N_VEH*F_IN = 256 = D_DIM.
__global__ void k_gcn(const float* __restrict__ traj, const float* __restrict__ gcn_w,
                      const float* __restrict__ gcn_b, _Float16* __restrict__ g_h) {
    __shared__ float s_tr[N_VEH * F_IN];
    __shared__ float s_xbar[F_IN];
    int bt  = blockIdx.x;          // bt = b*T + t  (b-major)
    int tid = threadIdx.x;         // 0..255
    s_tr[tid] = traj[(size_t)bt * (N_VEH * F_IN) + tid];
    __syncthreads();
    if (tid < F_IN) {
        float acc = 0.f;
        #pragma unroll
        for (int n = 0; n < N_VEH; n++) acc += s_tr[n * F_IN + tid];
        s_xbar[tid] = acc * (1.0f / N_VEH);
    }
    __syncthreads();
    float acc = gcn_b[tid];
    #pragma unroll
    for (int f = 0; f < F_IN; f++) acc += s_xbar[f] * gcn_w[f * D_DIM + tid];
    g_h[(size_t)bt * D_DIM + tid] = (_Float16)fmaxf(acc, 0.f);
}

// ---------------- WMMA GEMM: C = A(MxK,f16) @ W(NxK,f16)^T + bias ----------------
// One wave per 16x16 tile. perm: 0 = identity, 1 = out_row=(m%T)*B+m/T (b-major -> t-major),
// 2 = out_row=(m%B)*T+m/B (t-major -> b-major).
__global__ void k_gemm_wmma(const _Float16* __restrict__ A, const _Float16* __restrict__ W,
                            const float* __restrict__ bias, float* __restrict__ Cf,
                            _Float16* __restrict__ Ch, int M, int N, int K,
                            int relu, int accum, int perm) {
    int gw   = (blockIdx.x * blockDim.x + threadIdx.x) >> 5;  // global wave id
    int lane = threadIdx.x & 31;
    int Nt   = N >> 4;
    int total = (M >> 4) * Nt;
    if (gw >= total) return;                // wave-uniform: EXEC stays all-1 for WMMA
    int m0 = (gw / Nt) << 4;
    int n0 = (gw % Nt) << 4;
    int l  = lane & 15;
    int hi = lane >> 4;

    // A 16x32 f16 layout: lanes 0-15 = row m0+l with K=0..15 contiguous;
    //                     lanes 16-31 = row m0+l with K=16..31.
    const _Float16* Ap = A + (size_t)(m0 + l) * K + (hi ? 16 : 0);
    // B 32x16 f16 layout: lane l (0-15) holds column n0+l, K=0..15 contiguous;
    //                     lanes 16-31 hold column n0+l, K=16..31.
    const _Float16* Wp = W + (size_t)(n0 + l) * K + (hi ? 16 : 0);

    v8f acc = {};
    for (int k0 = 0; k0 < K; k0 += 32) {
        half8 a0 = *(const half8*)(Ap + k0);
        half8 a1 = *(const half8*)(Ap + k0 + 8);
        v16h a;
        #pragma unroll
        for (int i = 0; i < 8; i++) { a[i] = a0[i]; a[i + 8] = a1[i]; }
        v16h b = *(const v16h*)(Wp + k0);
        acc = __builtin_amdgcn_wmma_f32_16x16x32_f16(false, a, false, b,
                                                     (short)0, acc, false, false);
    }

    // C/D layout: VGPR r -> (M = m0 + r + 8*hi, N = n0 + l)
    int   nn = n0 + l;
    float bv = bias ? bias[nn] : 0.f;
    int   mb = m0 + hi * 8;
    #pragma unroll
    for (int r = 0; r < 8; r++) {
        int m   = mb + r;
        int row = m;
        if (perm == 1)      row = (m % T_DIM) * B_DIM + m / T_DIM;
        else if (perm == 2) row = (m % B_DIM) * T_DIM + m / B_DIM;
        size_t off = (size_t)row * N + nn;
        float v = acc[r] + bv;
        if (accum) v += Cf[off];
        if (relu)  v = fmaxf(v, 0.f);
        if (Cf) Cf[off] = v;
        if (Ch) Ch[off] = (_Float16)v;
    }
}

// ---------------- LSTM gates (torch order i,f,g,o) ----------------
__global__ void k_lstm_gates(const float* __restrict__ z, float* __restrict__ c,
                             _Float16* __restrict__ lstm_out_h, int t) {
    int i = blockIdx.x * blockDim.x + threadIdx.x;    // B*LSTM_HD
    if (i >= B_DIM * LSTM_HD) return;
    int b = i / LSTM_HD, j = i % LSTM_HD;
    const float* zb = z + (size_t)b * GATE4;
    float ig = zb[j];
    float fg = zb[LSTM_HD + j];
    float gg = zb[2 * LSTM_HD + j];
    float og = zb[3 * LSTM_HD + j];
    float si = 1.f / (1.f + expf(-ig));
    float sf = 1.f / (1.f + expf(-fg));
    float so = 1.f / (1.f + expf(-og));
    float cn = sf * c[i] + si * tanhf(gg);
    c[i] = cn;
    float h = so * tanhf(cn);
    lstm_out_h[((size_t)t * B_DIM + b) * LSTM_HD + j] = (_Float16)h;  // t-major rows
}

// ---------------- host ----------------
extern "C" void kernel_launch(void* const* d_in, const int* in_sizes, int n_in,
                              void* d_out, int out_size, void* d_ws, size_t ws_size,
                              hipStream_t stream) {
    const float* traj       = (const float*)d_in[0];
    const float* gcn_w      = (const float*)d_in[1];
    const float* gcn_b      = (const float*)d_in[2];
    const float* in_proj_w  = (const float*)d_in[3];
    const float* in_proj_b  = (const float*)d_in[4];
    const float* out_proj_w = (const float*)d_in[5];
    const float* out_proj_b = (const float*)d_in[6];
    // d_in[7]=va_w, d_in[8]=va_b: drop out exactly (softmax over identical logits is uniform)
    const float* w_ih   = (const float*)d_in[9];
    const float* w_hh   = (const float*)d_in[10];
    const float* b_ih   = (const float*)d_in[11];
    const float* b_hh   = (const float*)d_in[12];
    const float* lin1_w = (const float*)d_in[13];
    const float* lin1_b = (const float*)d_in[14];
    const float* lin2_w = (const float*)d_in[15];
    const float* lin2_b = (const float*)d_in[16];
    float* out = (float*)d_out;

    char* ws = (char*)d_ws;
    size_t off = 0;
    auto alloc = [&](size_t bytes) -> char* {
        char* p = ws + off;
        off += (bytes + 255) & ~(size_t)255;
        return p;
    };

    const int BT = B_DIM * T_DIM;  // 3200
    _Float16* wv_h   = (_Float16*)alloc((size_t)D_DIM * D_DIM * 2);
    _Float16* wout_h = (_Float16*)alloc((size_t)D_DIM * D_DIM * 2);
    _Float16* wih_h  = (_Float16*)alloc((size_t)GATE4 * D_DIM * 2);
    _Float16* whh_h  = (_Float16*)alloc((size_t)GATE4 * LSTM_HD * 2);
    _Float16* l1w_h  = (_Float16*)alloc((size_t)ATT_OUT * LSTM_HD * 2);
    _Float16* l2w_h  = (_Float16*)alloc((size_t)LATENT * ATT_OUT * 2);
    float*    bsum   = (float*)   alloc((size_t)GATE4 * 4);
    // zero-init region: c state (f32) + h0 (f16), contiguous
    char*     zreg   = alloc((size_t)B_DIM * LSTM_HD * 4 + (size_t)B_DIM * LSTM_HD * 2);
    float*    cstate = (float*)zreg;
    _Float16* h0_h   = (_Float16*)(zreg + (size_t)B_DIM * LSTM_HD * 4);
    _Float16* g_h    = (_Float16*)alloc((size_t)BT * D_DIM * 2);
    _Float16* v_h    = (_Float16*)alloc((size_t)BT * D_DIM * 2);
    _Float16* agg_h  = (_Float16*)alloc((size_t)BT * D_DIM * 2);   // t-major rows
    float*    zbuf   = (float*)   alloc((size_t)B_DIM * GATE4 * 4);
    _Float16* lstm_h = (_Float16*)alloc((size_t)BT * LSTM_HD * 2); // t-major rows
    _Float16* enc1_h = (_Float16*)alloc((size_t)BT * ATT_OUT * 2); // t-major rows

    auto conv = [&](const float* s, _Float16* d, int n) {
        k_f32_to_f16<<<(n + 255) / 256, 256, 0, stream>>>(s, d, n);
    };
    auto gemm = [&](const _Float16* A, const _Float16* W, const float* bias,
                    float* Cf, _Float16* Ch, int M, int N, int K,
                    int relu, int accum, int perm) {
        int waves  = (M >> 4) * (N >> 4);
        int blocks = (waves + 7) / 8;   // 8 waves (256 threads) per block
        k_gemm_wmma<<<blocks, 256, 0, stream>>>(A, W, bias, Cf, Ch, M, N, K, relu, accum, perm);
    };

    // 0) zero LSTM state (c, h0) — done every call (ws is poisoned once, never restored)
    {
        int words = (B_DIM * LSTM_HD * 4 + B_DIM * LSTM_HD * 2) / 4;
        k_zero_u32<<<(words + 255) / 256, 256, 0, stream>>>((uint32_t*)zreg, words);
    }
    // 1) weight conversions (f32 -> f16); V block of in_proj only
    conv(in_proj_w + 2 * D_DIM * D_DIM, wv_h, D_DIM * D_DIM);
    conv(out_proj_w, wout_h, D_DIM * D_DIM);
    conv(w_ih,   wih_h, GATE4 * D_DIM);
    conv(w_hh,   whh_h, GATE4 * LSTM_HD);
    conv(lin1_w, l1w_h, ATT_OUT * LSTM_HD);
    conv(lin2_w, l2w_h, LATENT * ATT_OUT);
    k_bias_sum<<<(GATE4 + 255) / 256, 256, 0, stream>>>(b_ih, b_hh, bsum, GATE4);

    // 2) collapsed GCN: g_row (b-major rows), f16
    k_gcn<<<BT, 256, 0, stream>>>(traj, gcn_w, gcn_b, g_h);

    // 3) collapsed attention: agg = out_proj(v_proj(g_row)); store agg t-major
    gemm(g_h, wv_h, in_proj_b + 2 * D_DIM, nullptr, v_h, BT, D_DIM, D_DIM, 0, 0, 0);
    gemm(v_h, wout_h, out_proj_b, nullptr, agg_h, BT, D_DIM, D_DIM, 0, 0, 1);

    // 4) LSTM over T (sequential); x_t = agg rows [t*B, t*B+32)
    for (int t = 0; t < T_DIM; t++) {
        const _Float16* xt = agg_h + (size_t)t * B_DIM * D_DIM;
        const _Float16* hp = (t == 0) ? h0_h : (lstm_h + (size_t)(t - 1) * B_DIM * LSTM_HD);
        gemm(xt, wih_h, bsum, zbuf, nullptr, B_DIM, GATE4, D_DIM, 0, 0, 0);      // z  = x@Wih^T + (b_ih+b_hh)
        gemm(hp, whh_h, nullptr, zbuf, nullptr, B_DIM, GATE4, LSTM_HD, 0, 1, 0); // z += h@Whh^T
        k_lstm_gates<<<(B_DIM * LSTM_HD + 255) / 256, 256, 0, stream>>>(zbuf, cstate, lstm_h, t);
    }

    // 5) MLP head: relu(lstm @ lin1^T + b1) @ lin2^T + b2; final store back to b-major (B,T,LATENT)
    gemm(lstm_h, l1w_h, lin1_b, nullptr, enc1_h, BT, ATT_OUT, LSTM_HD, 1, 0, 0);
    gemm(enc1_h, l2w_h, lin2_b, out, nullptr, BT, LATENT, ATT_OUT, 0, 0, 2);
}
